// GATSepEncoder_17042430231191
// MI455X (gfx1250) — matrix operand order repaired and checked by reading, not compiled, not run on hardware
//
#include <hip/hip_runtime.h>
#include <hip/hip_bf16.h>

// ---------------------------------------------------------------------------
// GATSepEncoder for MI455X (gfx1250, wave32, WMMA + async-to-LDS)
//   h = gelu(x @ W_in + b_in)
//   per layer: hi = h@Wl+bl ; au/av = hi@Wau/Wav ; edge softmax over dst ;
//              msg = segsum(hi[src]*p) ; h += gelu([hi,msg]@Wf1+bf1)@Wf2+bf2
// All GEMM operands are bf16-resident in HBM; tiles are staged into LDS with
// GLOBAL_LOAD_ASYNC_TO_LDS_B128 (ASYNCcnt), double-buffered against
// v_wmma_f32_16x16x32_bf16. Block tile 64(M) x 128(N): full output width per
// block so each A row is streamed from HBM exactly once per GEMM.
// ---------------------------------------------------------------------------

typedef __attribute__((ext_vector_type(16))) __bf16 v16bf;
typedef __attribute__((ext_vector_type(8)))  float v8f;

#define LDS_STRIDE 40   // bf16 elems per LDS row: 80B -> 16B-aligned, bank-skewed

__device__ __forceinline__ float gelu_exact(float x) {
    return 0.5f * x * (1.0f + erff(x * 0.70710678118654752f));
}

__device__ __forceinline__ float atomicMaxFloat(float* addr, float value) {
    int* ai = (int*)addr;
    int old = __float_as_int(*addr);
    while (value > __int_as_float(old)) {
        int assumed = old;
        old = atomicCAS(ai, assumed, __float_as_int(value));
        if (old == assumed) break;
    }
    return __int_as_float(old);
}

// ---- CDNA5 async copy: 16 bytes / lane, HBM -> LDS (ASYNCcnt tracked) -----
__device__ __forceinline__ void async_copy_b128(unsigned lds_off, const void* gaddr) {
    asm volatile("global_load_async_to_lds_b128 %0, %1, off"
                 :: "v"(lds_off), "v"(gaddr)
                 : "memory");
}
__device__ __forceinline__ void wait_async_0() {
    asm volatile("s_wait_asynccnt 0x0" ::: "memory");
}
__device__ __forceinline__ void wait_async_3() {
    asm volatile("s_wait_asynccnt 0x3" ::: "memory");
}

// ---------------------------------------------------------------------------
// Tiled GEMM: C[M,128] = op(A[M,K] @ B[K,128] + bias) (+ residual R)
//   A  : bf16 row-major [M][lda]
//   BT : bf16 row-major [128][K]  (B transposed; tile copies are contiguous)
// Block = 256 threads = 8 waves (4 M-groups x 2 N-groups); each wave owns a
// 16x64 output patch = four 16x16x32 WMMA accumulators sharing one A frag.
// Per k-tile each thread issues 3 async b128 copies (1 A + 2 B halves);
// double buffering overlaps tile t+1 copies with tile t WMMAs.
// Epilogue writes f32 C and/or bf16 Cb (feeds the next GEMM's A).
// ---------------------------------------------------------------------------
template<bool GELU, bool BIAS, bool RESID, bool WF32, bool WBF16>
__global__ __launch_bounds__(256)
void gemm_async_wmma(const __bf16* __restrict__ A, int lda,
                     const __bf16* __restrict__ BT,
                     const float* __restrict__ bias,
                     float* __restrict__ C, int ldc,
                     __bf16* __restrict__ Cb, int ldcb,
                     const float* __restrict__ R, int ldr,
                     int M, int K)
{
    __shared__ __bf16 sA[2][64  * LDS_STRIDE];   // [row][k]  64x32 tile
    __shared__ __bf16 sB[2][128 * LDS_STRIDE];   // [col][k] 128x32 tile (from BT)

    const int tid  = threadIdx.x;
    const int lane = tid & 31;
    const int wave = tid >> 5;
    const int wm   = wave & 3;     // 0..3 -> 16-row group
    const int wn   = wave >> 2;    // 0..1 -> 64-col group

    const int rowBase = blockIdx.x * 64;

    // staging: A = 1 b128/thread, B = 2 b128/thread (cols 0..63 and 64..127)
    const int ar = tid >> 2;              // 0..63 tile row / B col group 0
    const int as = (tid & 3) << 3;        // 0,8,16,24 (bf16 elems = 16B)
    const int arClamp = (rowBase + ar < M) ? (rowBase + ar) : (M - 1);
    const __bf16* aSrc  = A  + (size_t)arClamp * lda + as;
    const __bf16* bSrc0 = BT + (size_t)ar        * K + as;
    const __bf16* bSrc1 = BT + (size_t)(ar + 64) * K + as;

    const unsigned ldsA0  = (unsigned)(size_t)&sA[0][ar * LDS_STRIDE + as];
    const unsigned ldsA1  = (unsigned)(size_t)&sA[1][ar * LDS_STRIDE + as];
    const unsigned ldsB00 = (unsigned)(size_t)&sB[0][ar * LDS_STRIDE + as];
    const unsigned ldsB01 = (unsigned)(size_t)&sB[1][ar * LDS_STRIDE + as];
    const unsigned ldsB10 = (unsigned)(size_t)&sB[0][(ar + 64) * LDS_STRIDE + as];
    const unsigned ldsB11 = (unsigned)(size_t)&sB[1][(ar + 64) * LDS_STRIDE + as];

    v8f acc0 = {}, acc1 = {}, acc2 = {}, acc3 = {};

    // fragment addressing (documented 16-bit WMMA VGPR layouts)
    const int arow = wm * 16 + (lane & 15);
    const int akb  = (lane < 16) ? 0 : 8;     // A: K 0..7&16..23 / 8..15&24..31
    const int bkb  = (lane < 16) ? 0 : 8;     // B: K 0..7&16..23 / 8..15&24..31
    const int bcol = wn * 64 + (lane & 15);

    // prologue: issue tile 0 (3 async ops)
    async_copy_b128(ldsA0,  aSrc);
    async_copy_b128(ldsB00, bSrc0);
    async_copy_b128(ldsB10, bSrc1);

    const int nt = K >> 5;
    for (int t = 0; t < nt; ++t) {
        const int cur = t & 1;
        if (t + 1 < nt) {               // issue next tile, then wait current
            const int koff = (t + 1) << 5;
            async_copy_b128(cur ? ldsA0  : ldsA1,  aSrc  + koff);
            async_copy_b128(cur ? ldsB00 : ldsB01, bSrc0 + koff);
            async_copy_b128(cur ? ldsB10 : ldsB11, bSrc1 + koff);
            wait_async_3();
        } else {
            wait_async_0();
        }
        __syncthreads();

        const __bf16* pa = &sA[cur][arow * LDS_STRIDE];
        const __bf16* pb = &sB[cur][bcol * LDS_STRIDE];
        union { v16bf v; uint4 q[2]; } fa, fb0, fb1, fb2, fb3;
        fa.q[0]  = *(const uint4*)(pa + akb);
        fa.q[1]  = *(const uint4*)(pa + akb + 16);
        fb0.q[0] = *(const uint4*)(pb + bkb);
        fb0.q[1] = *(const uint4*)(pb + bkb + 16);
        fb1.q[0] = *(const uint4*)(pb + 16 * LDS_STRIDE + bkb);
        fb1.q[1] = *(const uint4*)(pb + 16 * LDS_STRIDE + bkb + 16);
        fb2.q[0] = *(const uint4*)(pb + 32 * LDS_STRIDE + bkb);
        fb2.q[1] = *(const uint4*)(pb + 32 * LDS_STRIDE + bkb + 16);
        fb3.q[0] = *(const uint4*)(pb + 48 * LDS_STRIDE + bkb);
        fb3.q[1] = *(const uint4*)(pb + 48 * LDS_STRIDE + bkb + 16);

        acc0 = __builtin_amdgcn_wmma_f32_16x16x32_bf16(false, fa.v, false, fb0.v,
                                                       (short)0, acc0, false, false);
        acc1 = __builtin_amdgcn_wmma_f32_16x16x32_bf16(false, fa.v, false, fb1.v,
                                                       (short)0, acc1, false, false);
        acc2 = __builtin_amdgcn_wmma_f32_16x16x32_bf16(false, fa.v, false, fb2.v,
                                                       (short)0, acc2, false, false);
        acc3 = __builtin_amdgcn_wmma_f32_16x16x32_bf16(false, fa.v, false, fb3.v,
                                                       (short)0, acc3, false, false);
        __syncthreads();                 // before next iteration overwrites cur
    }

    // ---- epilogue: C/D layout lane<16 -> N=lane, M=r ; lane>=16 -> M=r+8 ----
    const int mrow = rowBase + wm * 16 + ((lane >= 16) ? 8 : 0);
    const int col0 = wn * 64 + (lane & 15);
    float b[4] = {0.f, 0.f, 0.f, 0.f};
    if (BIAS) {
        #pragma unroll
        for (int c = 0; c < 4; ++c) b[c] = bias[col0 + 16 * c];
    }
    #pragma unroll
    for (int r = 0; r < 8; ++r) {
        int rr = mrow + r;
        if (rr < M) {
            float v[4] = { acc0[r] + b[0], acc1[r] + b[1],
                           acc2[r] + b[2], acc3[r] + b[3] };
            #pragma unroll
            for (int c = 0; c < 4; ++c) {
                int cc = col0 + 16 * c;
                float vv = v[c];
                if (GELU)  vv = gelu_exact(vv);
                if (RESID) vv += R[(size_t)rr * ldr + cc];
                if (WF32)  C[(size_t)rr * ldc + cc] = vv;
                if (WBF16) Cb[(size_t)rr * ldcb + cc] = (__bf16)vv;
            }
        }
    }
}

// ---------------------------------------------------------------------------
// Prep: f32 -> bf16 (same layout), 2 elems/thread (v_cvt_pk_bf16_f32)
// ---------------------------------------------------------------------------
__global__ __launch_bounds__(256)
void cvt_bf16_kernel(const float* __restrict__ in, __bf16* __restrict__ out, long n)
{
    long i = ((long)blockIdx.x * 256 + threadIdx.x) * 2;
    if (i + 1 < n) {
        float2 v = *(const float2*)(in + i);
        union { __bf16 b[2]; unsigned u; } pk;
        pk.b[0] = (__bf16)v.x;
        pk.b[1] = (__bf16)v.y;
        *(unsigned*)(out + i) = pk.u;
    } else if (i < n) {
        out[i] = (__bf16)in[i];
    }
}

// Prep: f32 [K][N] -> bf16 [N][K] (B-transposed layout for GEMM)
__global__ __launch_bounds__(256)
void transpose_cvt_kernel(const float* __restrict__ in, __bf16* __restrict__ out,
                          int K, int N)
{
    int i = blockIdx.x * 256 + threadIdx.x;
    if (i >= K * N) return;
    int n = i / K, k = i - n * K;
    out[i] = (__bf16)in[(size_t)k * N + n];
}

// ---------------------------------------------------------------------------
// au = hi @ Wau + bau, av = hi @ Wav        (D=128, H=4; warp per node)
// ---------------------------------------------------------------------------
__global__ __launch_bounds__(256)
void attn_proj_kernel(const float* __restrict__ hi, int ldh,
                      const float* __restrict__ Wau, const float* __restrict__ bau,
                      const float* __restrict__ Wav,
                      float* __restrict__ au, float* __restrict__ av, int Nn)
{
    int w    = (blockIdx.x * blockDim.x + threadIdx.x) >> 5;
    int lane = threadIdx.x & 31;
    if (w >= Nn) return;
    const float* hrow = hi + (size_t)w * ldh;
    float x[4];
    #pragma unroll
    for (int j = 0; j < 4; ++j) x[j] = hrow[lane * 4 + j];
    #pragma unroll
    for (int h = 0; h < 4; ++h) {
        float su = 0.f, sv = 0.f;
        #pragma unroll
        for (int j = 0; j < 4; ++j) {
            int k = lane * 4 + j;
            su += x[j] * Wau[k * 4 + h];
            sv += x[j] * Wav[k * 4 + h];
        }
        #pragma unroll
        for (int off = 16; off > 0; off >>= 1) {
            su += __shfl_xor(su, off, 32);
            sv += __shfl_xor(sv, off, 32);
        }
        if (lane == 0) {
            au[w * 4 + h] = su + bau[h];
            av[w * 4 + h] = sv;
        }
    }
}

// Per-layer init: zero msg half of comb, smax=-inf, z=0
__global__ __launch_bounds__(256)
void layer_init_kernel(float* __restrict__ comb, float* __restrict__ smax,
                       float* __restrict__ z, int Nn)
{
    int i = blockIdx.x * 256 + threadIdx.x;
    if (i < Nn * 128) comb[(size_t)(i >> 7) * 256 + 128 + (i & 127)] = 0.f;
    if (i < Nn * 4) { smax[i] = -3.0e38f; z[i] = 0.f; }
}

// Edge pass 1: s = leakyrelu(au[src]+av[dst]); segment-max into smax[dst]
__global__ __launch_bounds__(256)
void edge_score_kernel(const int* __restrict__ src, const int* __restrict__ dst,
                       const float* __restrict__ au, const float* __restrict__ av,
                       float* __restrict__ sbuf, float* __restrict__ smax, int E)
{
    int t = blockIdx.x * 256 + threadIdx.x;
    if (t >= E * 4) return;
    int e = t >> 2, h = t & 3;
    int s = src[e], d = dst[e];
    float v = au[s * 4 + h] + av[d * 4 + h];
    v = (v > 0.f) ? v : 0.2f * v;
    sbuf[t] = v;
    atomicMaxFloat(&smax[d * 4 + h], v);
}

// Edge pass 2: e = exp(s - smax[dst]); segment-sum into z[dst]
__global__ __launch_bounds__(256)
void edge_exp_kernel(const int* __restrict__ dst,
                     float* __restrict__ sbuf, const float* __restrict__ smax,
                     float* __restrict__ z, int E)
{
    int t = blockIdx.x * 256 + threadIdx.x;
    if (t >= E * 4) return;
    int e = t >> 2, h = t & 3;
    int d = dst[e];
    float ev = expf(sbuf[t] - smax[d * 4 + h]);
    sbuf[t] = ev;
    atomicAdd(&z[d * 4 + h], ev);
}

// Edge pass 3: msg[dst, i] += hi[src, i] * p[head(i)]   (warp per edge)
// hi = comb[:, 0:128], msg = comb[:, 128:256]; head(i) = i & 3
__global__ __launch_bounds__(256)
void edge_msg_kernel(const int* __restrict__ src, const int* __restrict__ dst,
                     const float* __restrict__ sbuf, const float* __restrict__ z,
                     float* __restrict__ comb, int E)
{
    int w    = (blockIdx.x * 256 + threadIdx.x) >> 5;
    int lane = threadIdx.x & 31;
    if (w >= E) return;
    int s = src[w], d = dst[w];
    float p[4];
    #pragma unroll
    for (int j = 0; j < 4; ++j) p[j] = sbuf[w * 4 + j] / z[d * 4 + j];
    const float* hrow = comb + (size_t)s * 256;
    float*       mrow = comb + (size_t)d * 256 + 128;
    #pragma unroll
    for (int j = 0; j < 4; ++j) {
        int c = lane * 4 + j;                       // head = c & 3 = j
        atomicAdd(&mrow[c], hrow[c] * p[j]);
    }
}

// ---------------------------------------------------------------------------
extern "C" void kernel_launch(void* const* d_in, const int* in_sizes, int n_in,
                              void* d_out, int out_size, void* d_ws, size_t ws_size,
                              hipStream_t stream)
{
    const float* x    = (const float*)d_in[0];
    const int*   esrc = (const int*)  d_in[1];
    const int*   edst = (const int*)  d_in[2];
    const float* W_in = (const float*)d_in[3];
    const float* b_in = (const float*)d_in[4];
    const float* Wl   = (const float*)d_in[5];   // [L,128,128]
    const float* bl   = (const float*)d_in[6];   // [L,128]
    const float* Wau  = (const float*)d_in[7];   // [L,128,4]
    const float* bau  = (const float*)d_in[8];   // [L,4]
    const float* Wav  = (const float*)d_in[9];   // [L,128,4]
    const float* Wf1  = (const float*)d_in[10];  // [L,256,128]
    const float* bf1  = (const float*)d_in[11];  // [L,128]
    const float* Wf2  = (const float*)d_in[12];  // [L,128,128]
    const float* bf2  = (const float*)d_in[13];  // [L,128]

    const int Nn = in_sizes[0] / 256;            // 50000
    const int E  = in_sizes[1];                  // 800000
    const int L  = in_sizes[5] / (128 * 128);    // 2

    float* h = (float*)d_out;                    // [N,128], final output

    // workspace layout
    char* ws = (char*)d_ws;
    size_t off = 0;
    auto walloc = [&](size_t bytes) -> void* {
        void* p = ws + off;
        off += (bytes + 255) & ~(size_t)255;
        return p;
    };
    float*  comb  = (float*) walloc((size_t)Nn * 256 * 4); // [N,2D] f32: hi | msg
    float*  au    = (float*) walloc((size_t)Nn * 4 * 4);
    float*  av    = (float*) walloc((size_t)Nn * 4 * 4);
    float*  sbuf  = (float*) walloc((size_t)E  * 4 * 4);
    float*  smax  = (float*) walloc((size_t)Nn * 4 * 4);
    float*  zbuf  = (float*) walloc((size_t)Nn * 4 * 4);
    __bf16* xb    = (__bf16*)walloc((size_t)Nn * 256 * 2); // bf16 activations
    __bf16* hb    = (__bf16*)walloc((size_t)Nn * 128 * 2);
    __bf16* combb = (__bf16*)walloc((size_t)Nn * 256 * 2);
    __bf16* tb    = (__bf16*)walloc((size_t)Nn * 128 * 2);
    __bf16* WinT  = (__bf16*)walloc((size_t)128 * 256 * 2); // bf16 [N][K] weights
    __bf16* WlT   = (__bf16*)walloc((size_t)L * 128 * 128 * 2);
    __bf16* Wf1T  = (__bf16*)walloc((size_t)L * 128 * 256 * 2);
    __bf16* Wf2T  = (__bf16*)walloc((size_t)L * 128 * 128 * 2);

    const dim3 blk(256);
    const dim3 gemm_grid((Nn + 63) / 64);        // full N=128 width per block

    // ---- prep: bf16-resident operands ----
    cvt_bf16_kernel<<<((size_t)Nn * 256 / 2 + 255) / 256, blk, 0, stream>>>(
        x, xb, (long)Nn * 256);
    transpose_cvt_kernel<<<(256 * 128 + 255) / 256, blk, 0, stream>>>(
        W_in, WinT, 256, 128);
    for (int l = 0; l < L; ++l) {
        transpose_cvt_kernel<<<(128 * 128 + 255) / 256, blk, 0, stream>>>(
            Wl + (size_t)l * 128 * 128, WlT + (size_t)l * 128 * 128, 128, 128);
        transpose_cvt_kernel<<<(256 * 128 + 255) / 256, blk, 0, stream>>>(
            Wf1 + (size_t)l * 256 * 128, Wf1T + (size_t)l * 128 * 256, 256, 128);
        transpose_cvt_kernel<<<(128 * 128 + 255) / 256, blk, 0, stream>>>(
            Wf2 + (size_t)l * 128 * 128, Wf2T + (size_t)l * 128 * 128, 128, 128);
    }

    // h = gelu(x @ W_in + b_in)   (writes f32 h and bf16 hb)
    gemm_async_wmma<true, true, false, true, true><<<gemm_grid, blk, 0, stream>>>(
        xb, 256, WinT, b_in, h, 128, hb, 128, nullptr, 0, Nn, 256);

    for (int l = 0; l < L; ++l) {
        layer_init_kernel<<<(Nn * 128 + 255) / 256, blk, 0, stream>>>(
            comb, smax, zbuf, Nn);

        // hi = h @ Wl + bl   -> comb[:, 0:128] (f32)
        gemm_async_wmma<false, true, false, true, false><<<gemm_grid, blk, 0, stream>>>(
            hb, 128, WlT + (size_t)l * 128 * 128, bl + l * 128,
            comb, 256, nullptr, 0, nullptr, 0, Nn, 128);

        // au/av projections
        attn_proj_kernel<<<(Nn * 32 + 255) / 256, blk, 0, stream>>>(
            comb, 256, Wau + (size_t)l * 128 * 4, bau + l * 4,
            Wav + (size_t)l * 128 * 4, au, av, Nn);

        // edge softmax (3 streaming passes, L2-resident atomics)
        edge_score_kernel<<<(E * 4 + 255) / 256, blk, 0, stream>>>(
            esrc, edst, au, av, sbuf, smax, E);
        edge_exp_kernel<<<(E * 4 + 255) / 256, blk, 0, stream>>>(
            edst, sbuf, smax, zbuf, E);
        edge_msg_kernel<<<(E * 32 + 255) / 256, blk, 0, stream>>>(
            esrc, edst, sbuf, zbuf, comb, E);

        // comb (f32) -> combb (bf16) for the FFN GEMM
        cvt_bf16_kernel<<<((size_t)Nn * 256 / 2 + 255) / 256, blk, 0, stream>>>(
            comb, combb, (long)Nn * 256);

        // t = gelu(comb @ Wf1 + bf1)   (bf16 only)
        gemm_async_wmma<true, true, false, false, true><<<gemm_grid, blk, 0, stream>>>(
            combb, 256, Wf1T + (size_t)l * 128 * 256, bf1 + l * 128,
            nullptr, 0, tb, 128, nullptr, 0, Nn, 256);

        // h = h + (t @ Wf2 + bf2)   (writes f32 h and bf16 hb for next layer)
        gemm_async_wmma<false, true, true, true, true><<<gemm_grid, blk, 0, stream>>>(
            tb, 128, Wf2T + (size_t)l * 128 * 128, bf2 + l * 128,
            h, 128, hb, 128, h, 128, Nn, 128);
    }
    (void)n_in; (void)out_size; (void)ws_size;
}
